// MutiHeadSelfAttention_55490977464676
// MI455X (gfx1250) — matrix operand order repaired and checked by reading, not compiled, not run on hardware
//
#include <hip/hip_runtime.h>

// ---------------- problem constants ----------------
#define BN   2
#define SS   2048
#define DM   1024
#define HH   16
#define DHH  64
#define MM   (BN * SS)      // 4096 rows of the projection GEMMs

typedef __attribute__((ext_vector_type(16))) __bf16 v16bf;
typedef __attribute__((ext_vector_type(8)))  float  v8f;

union Pack32 {                 // 32 bytes: two 16B vector loads -> one WMMA operand
    uint4  q[2];
    v16bf  v;
};

static __device__ __forceinline__ unsigned short f2bf(float f) {
    unsigned u = __float_as_uint(f);
    u += 0x7FFFu + ((u >> 16) & 1u);       // round-to-nearest-even
    return (unsigned short)(u >> 16);
}

static __device__ __forceinline__ v8f wmma_bf16(v16bf a, v16bf b, v8f c) {
    // 8 args: (neg_a, A, neg_b, B, c_mod, C, reuse_a, reuse_b)
    return __builtin_amdgcn_wmma_f32_16x16x32_bf16(false, a, false, b, (short)0, c,
                                                   false, false);
}

// async global -> LDS, 16 bytes per lane; tracked by ASYNCcnt
static __device__ __forceinline__ void async_cp16(unsigned lds_off,
                                                  const unsigned short* g) {
    unsigned long long ga = (unsigned long long)(uintptr_t)g;
    asm volatile("global_load_async_to_lds_b128 %0, %1, off"
                 :: "v"(lds_off), "v"(ga) : "memory");
}

// ---------------- kernel 1: x (f32) -> xb (bf16) ----------------
__global__ void k_cvt_x(const float* __restrict__ x, unsigned short* __restrict__ xb, int n) {
    int i = blockIdx.x * blockDim.x + threadIdx.x;
    if (i < n) xb[i] = f2bf(x[i]);
}

// ---------------- kernel 2: W[k][n] f32 -> Wt[n][k] bf16 (4 matrices via grid.z) ----------
__global__ void k_cvt_w(const float* __restrict__ wq, const float* __restrict__ wk,
                        const float* __restrict__ wv, const float* __restrict__ wr,
                        unsigned short* __restrict__ wt) {
    int z = blockIdx.z;
    const float* w = (z == 0) ? wq : (z == 1) ? wk : (z == 2) ? wv : wr;
    int i = blockIdx.x * blockDim.x + threadIdx.x;     // 0 .. 1M-1
    int n = i >> 10;
    int k = i & 1023;
    wt[(size_t)z * (DM * DM) + n * DM + k] = f2bf(w[k * DM + n]);
}

// ---------------- kernel 3: LDS-tiled projection GEMM -------------------------------------
// Block = 256 threads (8 waves) computes a 128x128 tile of Y = xb @ Wt^T (+bias).
// K stepped by 32 with double-buffered 128x32 LDS panels filled by async global->LDS copies.
// Each wave computes a 32x64 register sub-tile (2x4 accumulators, 8 WMMAs / K-step).
// z = 0:Q (scaled 1/sqrt(DH) -> [B,H,S,DH] bf16)  1:K (-> [B,H,S,DH] bf16)
// z = 2:V (-> [B,H,DH,S] bf16, transposed)         3:R (f32 -> d_out)
__global__ void __launch_bounds__(256)
k_gemm_proj(const unsigned short* __restrict__ xb,
            const unsigned short* __restrict__ wt,
            const float* __restrict__ bq, const float* __restrict__ bk,
            const float* __restrict__ bv, const float* __restrict__ br,
            unsigned short* __restrict__ qb,
            unsigned short* __restrict__ kbuf,
            unsigned short* __restrict__ vtb,
            float* __restrict__ outr) {
    __shared__ __attribute__((aligned(16))) unsigned short lA[2 * 128 * 32]; // 16 KB
    __shared__ __attribute__((aligned(16))) unsigned short lB[2 * 128 * 32]; // 16 KB

    const int z    = blockIdx.z;
    const int tid  = threadIdx.x;
    const int lane = tid & 31;
    const int wid  = tid >> 5;
    const int half = lane >> 4;
    const int l16  = lane & 15;

    const int bm = blockIdx.x >> 3;                    // 0..31  (M / 128)
    const int bn = blockIdx.x & 7;                     // 0..7   (N / 128)
    const int m0 = bm * 128;
    const int n0 = bn * 128;
    const int wm = wid & 3;                            // wave's 32-row quadrant
    const int wn = wid >> 2;                           // wave's 64-col half

    const unsigned short* wtz = wt + (size_t)z * (DM * DM);

    // per-thread async copy assignment: 2 chunks (16B) of A and of B per K-step
    const int row0 = tid >> 2;                         // chunks 0..255  -> rows 0..63
    const int seg0 = tid & 3;
    const int row1 = (tid + 256) >> 2;                 // chunks 256..511 -> rows 64..127
    const int seg1 = (tid + 256) & 3;
    const unsigned short* gA0 = xb  + (size_t)(m0 + row0) * DM + seg0 * 8;
    const unsigned short* gA1 = xb  + (size_t)(m0 + row1) * DM + seg1 * 8;
    const unsigned short* gB0 = wtz + (size_t)(n0 + row0) * DM + seg0 * 8;
    const unsigned short* gB1 = wtz + (size_t)(n0 + row1) * DM + seg1 * 8;
    const unsigned baseA = (unsigned)(uintptr_t)&lA[0];
    const unsigned baseB = (unsigned)(uintptr_t)&lB[0];
    const unsigned offA0 = baseA + (unsigned)(row0 * 32 + seg0 * 8) * 2;
    const unsigned offA1 = baseA + (unsigned)(row1 * 32 + seg1 * 8) * 2;
    const unsigned offB0 = baseB + (unsigned)(row0 * 32 + seg0 * 8) * 2;
    const unsigned offB1 = baseB + (unsigned)(row1 * 32 + seg1 * 8) * 2;

    v8f acc[2][4];
#pragma unroll
    for (int mi = 0; mi < 2; ++mi)
#pragma unroll
        for (int ni = 0; ni < 4; ++ni)
            acc[mi][ni] = (v8f){0.f, 0.f, 0.f, 0.f, 0.f, 0.f, 0.f, 0.f};

    const int KSTEPS = DM / 32;                        // 32

    // prologue: fill buffer 0
    async_cp16(offA0, gA0);
    async_cp16(offA1, gA1);
    async_cp16(offB0, gB0);
    async_cp16(offB1, gB1);

    for (int s = 0; s < KSTEPS; ++s) {
        const int cur = s & 1;
        if (s + 1 < KSTEPS) {
            const int kk  = (s + 1) * 32;
            const unsigned bofs = (unsigned)(((s + 1) & 1) * 128 * 32 * 2);
            async_cp16(offA0 + bofs, gA0 + kk);
            async_cp16(offA1 + bofs, gA1 + kk);
            async_cp16(offB0 + bofs, gB0 + kk);
            async_cp16(offB1 + bofs, gB1 + kk);
            asm volatile("s_wait_asynccnt 0x4" ::: "memory"); // batch s done
        } else {
            asm volatile("s_wait_asynccnt 0x0" ::: "memory");
        }
        __syncthreads();                               // panels for step s visible to all

        const unsigned short* pa = &lA[cur * 128 * 32];
        const unsigned short* pb = &lB[cur * 128 * 32];
        Pack32 a[2], b[4];
#pragma unroll
        for (int mi = 0; mi < 2; ++mi) {
            const unsigned short* ar = pa + (wm * 32 + mi * 16 + l16) * 32;
            a[mi].q[0] = *(const uint4*)(ar + half * 8);
            a[mi].q[1] = *(const uint4*)(ar + 16 + half * 8);
        }
#pragma unroll
        for (int ni = 0; ni < 4; ++ni) {
            const unsigned short* brp = pb + (wn * 64 + ni * 16 + l16) * 32;
            b[ni].q[0] = *(const uint4*)(brp + half * 8);
            b[ni].q[1] = *(const uint4*)(brp + 16 + half * 8);
        }
#pragma unroll
        for (int mi = 0; mi < 2; ++mi)
#pragma unroll
            for (int ni = 0; ni < 4; ++ni)
                acc[mi][ni] = wmma_bf16(a[mi].v, b[ni].v, acc[mi][ni]);

        __syncthreads();                               // done reading before refill
    }

    // ---------------- epilogue ----------------
    const float* bias = (z == 0) ? bq : (z == 1) ? bk : (z == 2) ? bv : br;
#pragma unroll
    for (int ni = 0; ni < 4; ++ni) {
        const int   col = n0 + wn * 64 + ni * 16 + l16; // C layout: lane = column
        const float bc  = bias[col];
        const int   h   = col >> 6;
        const int   dh  = col & 63;
#pragma unroll
        for (int mi = 0; mi < 2; ++mi) {
#pragma unroll
            for (int r = 0; r < 8; ++r) {
                const int   m   = m0 + wm * 32 + mi * 16 + r + half * 8;
                const float val = acc[mi][ni][r] + bc;
                const int   bb  = m >> 11;             // m / S
                const int   sq  = m & (SS - 1);
                if (z == 0) {
                    qb[((size_t)(bb * HH + h) * SS + sq) * DHH + dh] = f2bf(val * 0.125f);
                } else if (z == 1) {
                    kbuf[((size_t)(bb * HH + h) * SS + sq) * DHH + dh] = f2bf(val);
                } else if (z == 2) {
                    vtb[((size_t)(bb * HH + h) * DHH + dh) * SS + sq] = f2bf(val);
                } else {
                    outr[(size_t)m * DM + col] = val;  // residual, overwrites d_out
                }
            }
        }
    }
}

// ---------------- kernel 4: flash attention, one wave per 16 query rows ----------------
__global__ void k_attn(const unsigned short* __restrict__ qb,
                       const unsigned short* __restrict__ kbuf,
                       const unsigned short* __restrict__ vtb,
                       const unsigned char* __restrict__ mask,
                       float* __restrict__ out) {
    __shared__ __attribute__((aligned(16))) unsigned short plds[4][16 * 32];

    const int wid  = threadIdx.x >> 5;
    const int lane = threadIdx.x & 31;
    const int l16  = lane & 15;
    const int half = lane >> 4;

    const int gw = blockIdx.x * 4 + wid;               // 0 .. 4095
    const int qt = gw & 127;                           // S/16 q-tiles
    const int bh = gw >> 7;                            // 0 .. 31
    const int b  = bh >> 4;
    const int h  = bh & 15;
    const int q0 = qt * 16;

    const unsigned short* Qp = qb   + (size_t)bh * SS * DHH;
    const unsigned short* Kp = kbuf + (size_t)bh * SS * DHH;
    const unsigned short* Vp = vtb  + (size_t)bh * DHH * SS;

    // Q A-tiles (d = 0..31 and 32..63), lane = query row
    const unsigned short* qbase = Qp + (size_t)(q0 + l16) * DHH;
    Pack32 aq0, aq1;
    aq0.q[0] = *(const uint4*)(qbase + half * 8);
    aq0.q[1] = *(const uint4*)(qbase + 16 + half * 8);
    aq1.q[0] = *(const uint4*)(qbase + 32 + half * 8);
    aq1.q[1] = *(const uint4*)(qbase + 48 + half * 8);

    float mrow[8], lrow[8];
#pragma unroll
    for (int r = 0; r < 8; ++r) { mrow[r] = -1e30f; lrow[r] = 0.f; }
    v8f O[4];
#pragma unroll
    for (int t = 0; t < 4; ++t) O[t] = (v8f){0.f, 0.f, 0.f, 0.f, 0.f, 0.f, 0.f, 0.f};

    const int qend = q0 + 15;
    for (int kb0 = 0; kb0 <= qend; kb0 += 32) {
        const int key0 = kb0 + l16;
        const int key1 = key0 + 16;
        const bool pad0 = mask[b * SS + key0] != 0;
        const bool pad1 = mask[b * SS + key1] != 0;

        __builtin_prefetch(Kp + (size_t)(kb0 + 32 + l16) * DHH, 0, 1);

        // K^T B-tiles: lane = key column, d = 8*half+0..7 and 16+8*half+0..7 per lane (+32 for bk*1)
        const unsigned short* k0b = Kp + (size_t)key0 * DHH;
        const unsigned short* k1b = Kp + (size_t)key1 * DHH;
        Pack32 bk00, bk01, bk10, bk11;
        bk00.q[0] = *(const uint4*)(k0b + half * 8);
        bk00.q[1] = *(const uint4*)(k0b + 16 + half * 8);
        bk01.q[0] = *(const uint4*)(k0b + 32 + half * 8);
        bk01.q[1] = *(const uint4*)(k0b + 48 + half * 8);
        bk10.q[0] = *(const uint4*)(k1b + half * 8);
        bk10.q[1] = *(const uint4*)(k1b + 16 + half * 8);
        bk11.q[0] = *(const uint4*)(k1b + 32 + half * 8);
        bk11.q[1] = *(const uint4*)(k1b + 48 + half * 8);

        v8f s0 = {0.f, 0.f, 0.f, 0.f, 0.f, 0.f, 0.f, 0.f};
        v8f s1 = {0.f, 0.f, 0.f, 0.f, 0.f, 0.f, 0.f, 0.f};
        s0 = wmma_bf16(aq0.v, bk00.v, s0);
        s0 = wmma_bf16(aq1.v, bk01.v, s0);
        s1 = wmma_bf16(aq0.v, bk10.v, s1);
        s1 = wmma_bf16(aq1.v, bk11.v, s1);

        // mask + online softmax + stage P (bf16) into LDS row-major [16][32]
#pragma unroll
        for (int r = 0; r < 8; ++r) {
            const int mq = q0 + r + half * 8;          // global query row of this element
            float v0 = (pad0 && key0 <= mq) ? s0[r] : -1e30f;
            float v1 = (pad1 && key1 <= mq) ? s1[r] : -1e30f;
            float t  = fmaxf(v0, v1);
            t = fmaxf(t, __shfl_xor(t, 1));
            t = fmaxf(t, __shfl_xor(t, 2));
            t = fmaxf(t, __shfl_xor(t, 4));
            t = fmaxf(t, __shfl_xor(t, 8));
            const float mnew  = fmaxf(mrow[r], t);
            const float scale = __expf(mrow[r] - mnew);
            mrow[r] = mnew;
            const float p0 = __expf(v0 - mnew);
            const float p1 = __expf(v1 - mnew);
            float ssum = p0 + p1;
            ssum += __shfl_xor(ssum, 1);
            ssum += __shfl_xor(ssum, 2);
            ssum += __shfl_xor(ssum, 4);
            ssum += __shfl_xor(ssum, 8);
            lrow[r] = lrow[r] * scale + ssum;
            O[0][r] *= scale; O[1][r] *= scale; O[2][r] *= scale; O[3][r] *= scale;
            plds[wid][(r + half * 8) * 32 + l16]      = f2bf(p0);
            plds[wid][(r + half * 8) * 32 + 16 + l16] = f2bf(p1);
        }

        // wave-local LDS: enforce store->load ordering before A-layout reload
        asm volatile("s_wait_dscnt 0x0" ::: "memory");

        Pack32 pa;
        const unsigned short* pm = &plds[wid][l16 * 32];
        pa.q[0] = *(const uint4*)(pm + half * 8);
        pa.q[1] = *(const uint4*)(pm + 16 + half * 8);

        // P @ V : V^T B-tiles, lane = d column, keys kb0+8*half+0..7 and kb0+16+8*half+0..7 per lane
#pragma unroll
        for (int dt = 0; dt < 4; ++dt) {
            const unsigned short* vb = Vp + (size_t)(dt * 16 + l16) * SS + kb0 + half * 8;
            Pack32 bv;
            bv.q[0] = *(const uint4*)(vb);
            bv.q[1] = *(const uint4*)(vb + 16);
            O[dt] = wmma_bf16(pa.v, bv.v, O[dt]);
        }
    }

    // epilogue: 1/rowsum, query padding mask, accumulate onto residual in d_out
#pragma unroll
    for (int r = 0; r < 8; ++r) {
        const int   q     = q0 + r + half * 8;
        const float denom = fmaxf(lrow[r], 1e-30f);
        const float qm    = (mask[b * SS + q] != 0) ? 1.0f : 0.0f;
        const float sc    = qm / denom;
        const size_t orow = ((size_t)(b * SS + q)) * DM + h * DHH;
        out[orow +  0 + l16] += O[0][r] * sc;
        out[orow + 16 + l16] += O[1][r] * sc;
        out[orow + 32 + l16] += O[2][r] * sc;
        out[orow + 48 + l16] += O[3][r] * sc;
    }
}

// ---------------- launch ----------------
extern "C" void kernel_launch(void* const* d_in, const int* in_sizes, int n_in,
                              void* d_out, int out_size, void* d_ws, size_t ws_size,
                              hipStream_t stream) {
    const float*         x    = (const float*)d_in[0];
    const unsigned char* mask = (const unsigned char*)d_in[1];  // jax bool -> 1 byte
    const float* Wq = (const float*)d_in[2];
    const float* bq = (const float*)d_in[3];
    const float* Wk = (const float*)d_in[4];
    const float* bk = (const float*)d_in[5];
    const float* Wv = (const float*)d_in[6];
    const float* bv = (const float*)d_in[7];
    const float* Wr = (const float*)d_in[8];
    const float* br = (const float*)d_in[9];
    float* out = (float*)d_out;

    char* ws = (char*)d_ws;
    unsigned short* xb   = (unsigned short*)(ws);                        // 8 MB
    unsigned short* wt   = (unsigned short*)(ws + (size_t)( 8u << 20));  // 8 MB (4x 2MB)
    unsigned short* qb   = (unsigned short*)(ws + (size_t)(16u << 20));  // 8 MB
    unsigned short* kbuf = (unsigned short*)(ws + (size_t)(24u << 20));  // 8 MB
    unsigned short* vtb  = (unsigned short*)(ws + (size_t)(32u << 20));  // 8 MB

    const int nx = MM * DM;                                      // 4M elems
    k_cvt_x<<<(nx + 255) / 256, 256, 0, stream>>>(x, xb, nx);
    k_cvt_w<<<dim3((DM * DM) / 256, 1, 4), 256, 0, stream>>>(Wq, Wk, Wv, Wr, wt);
    k_gemm_proj<<<dim3(256, 1, 4), 256, 0, stream>>>(xb, wt, bq, bk, bv, br,
                                                     qb, kbuf, vtb, out);
    k_attn<<<1024, 128, 0, stream>>>(qb, kbuf, vtb, mask, out);
}
